// MoELayer_11587821765273
// MI455X (gfx1250) — matrix operand order repaired and checked by reading, not compiled, not run on hardware
//
#include <hip/hip_runtime.h>
#include <hip/hip_bf16.h>

// ---------------- problem constants (from reference) ----------------
#define T_TOK 2048   // B*S = 2*1024 tokens
#define DDIM  1024   // model dim
#define HDIM  1408   // hidden dim
#define NEXP  8      // experts
#define NWAVES 8     // 256 threads / wave32

typedef __attribute__((ext_vector_type(16))) _Float16 v16h;
typedef __attribute__((ext_vector_type(8)))  float    v8f;
typedef __attribute__((ext_vector_type(4)))  unsigned u32x4;
typedef __attribute__((ext_vector_type(8)))  int      i32x8;
typedef __attribute__((ext_vector_type(4)))  int      i32x4;

// ======================================================================
// Kernel 1: router — logits, softmax, top-2 gate, per-expert token lists
// one wave (32 lanes) per token
// ======================================================================
__global__ __launch_bounds__(256) void moe_router_kernel(
    const float* __restrict__ x, const float* __restrict__ Wr,
    float* __restrict__ gate, int* __restrict__ counts, int* __restrict__ lists)
{
    const int wave = threadIdx.x >> 5;
    const int lane = threadIdx.x & 31;
    const int t = blockIdx.x * (256 / 32) + wave;
    if (t >= T_TOK) return;

    float acc[NEXP];
#pragma unroll
    for (int e = 0; e < NEXP; ++e) acc[e] = 0.f;

    const float* xr = x + (size_t)t * DDIM;
    for (int d = lane; d < DDIM; d += 32) {
        const float xv = xr[d];
        const float* wr = Wr + (size_t)d * NEXP;   // 8 contiguous floats per row
#pragma unroll
        for (int e = 0; e < NEXP; ++e) acc[e] += xv * wr[e];
    }
    // wave32 reduction
#pragma unroll
    for (int off = 16; off > 0; off >>= 1) {
#pragma unroll
        for (int e = 0; e < NEXP; ++e) acc[e] += __shfl_xor(acc[e], off, 32);
    }

    if (lane == 0) {
        float mx = acc[0];
#pragma unroll
        for (int e = 1; e < NEXP; ++e) mx = fmaxf(mx, acc[e]);
        float p[NEXP], s = 0.f;
#pragma unroll
        for (int e = 0; e < NEXP; ++e) { p[e] = __expf(acc[e] - mx); s += p[e]; }
        const float inv = __builtin_amdgcn_rcpf(s);
#pragma unroll
        for (int e = 0; e < NEXP; ++e) p[e] *= inv;

        // top-2 (first max wins ties, matching lax.top_k ordering)
        int e1 = 0;
#pragma unroll
        for (int e = 1; e < NEXP; ++e) if (p[e] > p[e1]) e1 = e;
        int e2 = (e1 == 0) ? 1 : 0;
#pragma unroll
        for (int e = 0; e < NEXP; ++e) if (e != e1 && p[e] > p[e2]) e2 = e;

        float* gr = gate + (size_t)t * NEXP;
#pragma unroll
        for (int e = 0; e < NEXP; ++e) gr[e] = 0.f;
        gr[e1] = p[e1];
        gr[e2] = p[e2];

        int pos1 = atomicAdd(&counts[e1], 1); lists[e1 * T_TOK + pos1] = t;
        int pos2 = atomicAdd(&counts[e2], 1); lists[e2 * T_TOK + pos2] = t;
    }
}

// ======================================================================
// Kernel 2: per-expert SwiGLU FFN on gathered 16-token tiles, f16 WMMA.
//   grid.x = token tile (max 128), grid.y = expert; 256 thr = 8 waves
//   X tile gathered by the Tensor Data Mover (gather-mode D#, 16 row idx).
//   B fragments built directly in registers (lane = column, 16 K-strided
//   loads, coalesced across lanes) -> no per-k LDS staging, no barriers.
//   LDS overlay: [0,64K) = Xf32 (TDM dest) then Gs f16; [64K,96K) = Xs f16
// ======================================================================
__global__ __launch_bounds__(256) void moe_ffn_kernel(
    const float* __restrict__ x,  const float* __restrict__ W1,
    const float* __restrict__ W2, const float* __restrict__ W3,
    const float* __restrict__ gate, const int* __restrict__ counts,
    const int* __restrict__ lists, float* __restrict__ out)
{
    const int e    = blockIdx.y;
    const int tile = blockIdx.x;
    const int cnt  = counts[e];
    if (tile * 16 >= cnt) return;            // uniform per-workgroup early exit

    extern __shared__ char smem_raw[];
    float*    Xf = (float*)smem_raw;                    // 16*1024 f32 (TDM dest)
    _Float16* Gs = (_Float16*)smem_raw;                 // 16*1408 f16 (overlays Xf)
    _Float16* Xs = (_Float16*)(smem_raw + 16 * DDIM * 4);      // 16*1024 f16
    int*   tok_s  = (int*)(smem_raw + 16 * DDIM * 4 + 16 * DDIM * 2);
    float* gate_s = (float*)(tok_s + 16);

    const int tid  = threadIdx.x;
    const int wave = tid >> 5;
    const int lane = tid & 31;
    const int r    = lane & 15;     // row (A/C) or column (B) within tile
    const int hi   = lane >> 4;     // half-wave select

    if (tid < 16) {
        const int idx = tile * 16 + tid;
        const int t = (idx < cnt) ? lists[e * T_TOK + idx] : -1;
        tok_s[tid]  = t;
        gate_s[tid] = (t >= 0) ? gate[(size_t)t * NEXP + e] : 0.f;
    }
    __syncthreads();

    // ---- TDM gather: 16 token rows (f32, 4KB each) global -> LDS ----------
    if (tid < 32) {
        int tk[16];
#pragma unroll
        for (int i = 0; i < 16; ++i) { int t = tok_s[i]; tk[i] = (t < 0) ? 0 : t; }

        const unsigned long long ga = (unsigned long long)(uintptr_t)x;
        const unsigned lds = (unsigned)(uintptr_t)Xf;   // low 32b of flat = LDS offset

        u32x4 g0;
        g0[0] = (unsigned)__builtin_amdgcn_readfirstlane(0x80000001u); // count=1|gather|16b idx
        g0[1] = (unsigned)__builtin_amdgcn_readfirstlane((int)lds);    // lds_addr
        g0[2] = (unsigned)__builtin_amdgcn_readfirstlane((int)(unsigned)ga);
        g0[3] = (unsigned)__builtin_amdgcn_readfirstlane(
                    (int)(((unsigned)(ga >> 32) & 0x01FFFFFFu) | (2u << 30))); // type=2

        i32x8 g1;
        g1[0] = __builtin_amdgcn_readfirstlane(2 << 16);      // data_size = 4B
        g1[1] = __builtin_amdgcn_readfirstlane(DDIM << 16);   // tensor_dim0 = 1024
        g1[2] = __builtin_amdgcn_readfirstlane(T_TOK << 16);  // tensor_dim1 = 2048
        g1[3] = __builtin_amdgcn_readfirstlane(DDIM << 16);   // tile_dim0  = 1024
        g1[4] = __builtin_amdgcn_readfirstlane(16);           // tile_dim1 = #indices
        g1[5] = __builtin_amdgcn_readfirstlane(DDIM);         // dim0_stride = 1024
        g1[6] = 0; g1[7] = 0;

        i32x4 g2, g3;                                         // 16-bit row indices
#pragma unroll
        for (int i = 0; i < 4; ++i) {
            g2[i] = __builtin_amdgcn_readfirstlane((tk[2*i+1] << 16) | tk[2*i]);
            g3[i] = __builtin_amdgcn_readfirstlane((tk[8+2*i+1] << 16) | tk[8+2*i]);
        }

        i32x8 gx = {};   // extra descriptor group in the 6-arg builtin (unused)
        __builtin_amdgcn_tensor_load_to_lds(g0, g1, g2, g3, gx, 0);
        __builtin_amdgcn_s_wait_tensorcnt(0);
    }
    __syncthreads();

    // ---- convert X tile to f16 (padded rows -> 0), frees Xf for Gs --------
    for (int i = tid; i < 16 * DDIM; i += 256) {
        const int m = i >> 10;
        Xs[i] = (tok_s[m] >= 0) ? (_Float16)Xf[i] : (_Float16)0.f;
    }
    __syncthreads();

    const float* W1e = W1 + (size_t)e * DDIM * HDIM;   // [D,H] row-major
    const float* W2e = W2 + (size_t)e * DDIM * HDIM;
    const float* W3e = W3 + (size_t)e * HDIM * DDIM;   // [H,D] row-major

    // ---------------- Phase 1: a = X@W2e, b = X@W1e, G = silu(a)*b ---------
    for (int nb = wave; nb < HDIM / 16; nb += NWAVES) {   // 88/8 = 11, uniform
        const int n0 = nb * 16;
        v8f acc_a = {}, acc_b = {};
        const float* b2base = W2e + n0 + r;   // this lane's weight column
        const float* b1base = W1e + n0 + r;
        for (int k0 = 0; k0 < DDIM; k0 += 32) {
            // A 16x32 f16 from LDS: lane row M=r; K = hi*16 .. +16
            union { v16h v; uint4 u[2]; } A;
            const _Float16* arow = Xs + r * DDIM + k0 + hi * 16;
            A.u[0] = *(const uint4*)(arow);
            A.u[1] = *(const uint4*)(arow + 8);
            // B 32x16 f16 built in registers: lane col N=r, K = hi*16 .. +16
            const size_t rowoff = (size_t)(k0 + hi * 16) * HDIM;
            __builtin_prefetch(b2base + rowoff + (size_t)32 * HDIM, 0, 1);
            __builtin_prefetch(b1base + rowoff + (size_t)32 * HDIM, 0, 1);
            union { v16h v; _Float16 h[16]; } B2, B1;
#pragma unroll
            for (int j = 0; j < 16; ++j) {
                B2.h[j] = (_Float16)b2base[rowoff + (size_t)j * HDIM];
                B1.h[j] = (_Float16)b1base[rowoff + (size_t)j * HDIM];
            }
            acc_a = __builtin_amdgcn_wmma_f32_16x16x32_f16(
                        false, A.v, false, B2.v, (short)0, acc_a, false, false);
            acc_b = __builtin_amdgcn_wmma_f32_16x16x32_f16(
                        false, A.v, false, B1.v, (short)0, acc_b, false, false);
        }
        // C layout: VGPR v -> row M = v + 8*hi, col N = r ; fast silu via v_rcp
#pragma unroll
        for (int v = 0; v < 8; ++v) {
            const float a   = acc_a[v];
            const float sig = __builtin_amdgcn_rcpf(1.f + __expf(-a));
            Gs[(v + hi * 8) * HDIM + n0 + r] = (_Float16)(a * sig * acc_b[v]);
        }
    }
    __syncthreads();

    // ---------------- Phase 2: y = G @ W3e, scaled by gate, atomic combine --
    for (int nb = wave; nb < DDIM / 16; nb += NWAVES) {   // 64/8 = 8, uniform
        const int n0 = nb * 16;
        v8f acc = {};
        const float* b3base = W3e + n0 + r;
        for (int k0 = 0; k0 < HDIM; k0 += 32) {           // 44 steps
            union { v16h v; uint4 u[2]; } A;
            const _Float16* arow = Gs + r * HDIM + k0 + hi * 16;
            A.u[0] = *(const uint4*)(arow);
            A.u[1] = *(const uint4*)(arow + 8);
            const size_t rowoff = (size_t)(k0 + hi * 16) * DDIM;
            __builtin_prefetch(b3base + rowoff + (size_t)32 * DDIM, 0, 1);
            union { v16h v; _Float16 h[16]; } B;
#pragma unroll
            for (int j = 0; j < 16; ++j)
                B.h[j] = (_Float16)b3base[rowoff + (size_t)j * DDIM];
            acc = __builtin_amdgcn_wmma_f32_16x16x32_f16(
                      false, A.v, false, B.v, (short)0, acc, false, false);
        }
#pragma unroll
        for (int v = 0; v < 8; ++v) {
            const int m = v + hi * 8;
            const int t = tok_s[m];
            if (t >= 0)
                atomicAdd(&out[(size_t)t * DDIM + n0 + r], gate_s[m] * acc[v]);
        }
    }
}

// ======================================================================
// host launch
// ======================================================================
extern "C" void kernel_launch(void* const* d_in, const int* in_sizes, int n_in,
                              void* d_out, int out_size, void* d_ws, size_t ws_size,
                              hipStream_t stream) {
    (void)in_sizes; (void)n_in; (void)out_size; (void)ws_size;
    const float* x  = (const float*)d_in[0];
    const float* Wr = (const float*)d_in[1];
    const float* W1 = (const float*)d_in[2];
    const float* W2 = (const float*)d_in[3];
    const float* W3 = (const float*)d_in[4];
    float* out = (float*)d_out;

    // workspace layout (~128 KB): counts | gate[T,E] | lists[E,T]
    char*  ws     = (char*)d_ws;
    int*   counts = (int*)ws;                                  // 8 ints (64B slot)
    float* gate   = (float*)(ws + 64);                         // T*E floats
    int*   lists  = (int*)(ws + 64 + (size_t)T_TOK * NEXP * 4);// E*T ints

    (void)hipMemsetAsync(counts, 0, 64, stream);
    (void)hipMemsetAsync(out, 0, (size_t)T_TOK * DDIM * sizeof(float), stream);

    moe_router_kernel<<<T_TOK / 8, 256, 0, stream>>>(x, Wr, gate, counts, lists);

    const size_t smem = (size_t)16 * DDIM * 4      // Xf32 (TDM dest) / Gs overlay
                      + (size_t)16 * DDIM * 2      // Xs f16
                      + 16 * sizeof(int) + 16 * sizeof(float); // ~96.1 KB
    (void)hipFuncSetAttribute(reinterpret_cast<const void*>(moe_ffn_kernel),
                              hipFuncAttributeMaxDynamicSharedMemorySize, (int)smem);

    dim3 grid(T_TOK / 16, NEXP);   // 128 tiles max per expert; early-exit past count
    moe_ffn_kernel<<<grid, 256, smem, stream>>>(x, W1, W2, W3, gate, counts, lists, out);
}
